// NodeGridInteraction_71184787964483
// MI455X (gfx1250) — matrix operand order repaired and checked by reading, not compiled, not run on hardware
//
#include <hip/hip_runtime.h>
#include <hip/hip_bf16.h>
#include <cstdint>

typedef __attribute__((ext_vector_type(8)))  _Float16 v8h;
typedef __attribute__((ext_vector_type(16))) _Float16 v16h;
typedef __attribute__((ext_vector_type(8)))  float    v8f;

#define DHW_LOG2 18
#define DHW (1 << DHW_LOG2)
#define GDIM 64

// ---- WMMA fragment helpers (ISA 7.12.2 layouts, wave32) -------------------
__device__ __forceinline__ v16h frag_cat(v8h lo, v8h hi) {
  return __builtin_shufflevector(lo, hi, 0,1,2,3,4,5,6,7,8,9,10,11,12,13,14,15);
}
__device__ __forceinline__ v16h lds_frag_a(const _Float16* s, int ld, int m0, int k0, int lane) {
  const _Float16* p = s + (m0 + (lane & 15)) * ld + k0 + ((lane >> 4) << 3);
  return frag_cat(*(const v8h*)p, *(const v8h*)(p + 16));
}
__device__ __forceinline__ v16h lds_frag_b(const _Float16* s, int ld, int n0, int k0, int lane) {
  const _Float16* p = s + (n0 + (lane & 15)) * ld + k0 + ((lane >> 4) << 3);
  return frag_cat(*(const v8h*)p, *(const v8h*)(p + 16));
}
#define WMMA_F16(a, b, c) \
  __builtin_amdgcn_wmma_f32_16x16x32_f16(false, (a), false, (b), (short)0, (c), false, false)

// fast silu: x * v_rcp_f32(1+exp(-x)) -- avoids IEEE div_scale/fma chain
__device__ __forceinline__ float silu_f(float v) {
  return v * __builtin_amdgcn_rcpf(1.f + __expf(-v));
}

// ---------------------------------------------------------------------------
__global__ void zero_kernel(float4* __restrict__ p, size_t n4) {
  for (size_t i = (size_t)blockIdx.x * blockDim.x + threadIdx.x; i < n4;
       i += (size_t)gridDim.x * blockDim.x)
    p[i] = make_float4(0.f, 0.f, 0.f, 0.f);
}

// ---------------------------------------------------------------------------
// Per 64-node workgroup: gather + MLP + LN + projection + scatter.
__global__ void __launch_bounds__(256) node_kernel(
    const float* __restrict__ h, const float* __restrict__ x,
    const float* __restrict__ grid, const int* __restrict__ nbatch,
    const float* __restrict__ cmin, const float* __restrict__ sstr,
    const float* __restrict__ w1, const float* __restrict__ b1,
    const float* __restrict__ w2, const float* __restrict__ b2,
    const float* __restrict__ ln_g, const float* __restrict__ ln_b,
    const float* __restrict__ wn2g, const float* __restrict__ bn2g,
    float* __restrict__ hn_out, float* __restrict__ vol, int N)
{
  __shared__ __align__(16) union {
    _Float16 zs[64][200];          // z = [h | sampled], 192 used
    float    hn32[64][132];        // pre-LN fp32, 128 used (after GEMM1)
  } zb;
  __shared__ __align__(16) union {
    _Float16 w1s[128][200];        // GEMM1 weights (192 used)
    struct { _Float16 w2s[128][136]; _Float16 wgs[64][136]; } p2;  // later
  } wb;
  __shared__ __align__(16) _Float16 ms[64][136];    // silu(fc1) f16
  __shared__ __align__(16) _Float16 hnh[64][136];   // normalized hn f16
  __shared__ __align__(16) float projs[64][68];     // projection fp32
  __shared__ float red[64][8];

  const int tid  = threadIdx.x;
  const int lane = tid & 31, wv = tid >> 5;
  const int node0 = blockIdx.x * 64;

  // prefetch later-stage weights into cache (global_prefetch_b8) while we
  // do the scattered trilinear gather
  for (int i = tid * 32; i < 128 * 128; i += 256 * 32) __builtin_prefetch(&w2[i], 0, 0);
  for (int i = tid * 32; i < 64 * 128; i += 256 * 32)  __builtin_prefetch(&wn2g[i], 0, 0);

  // ---- trilinear corner data: 4 threads per node, kept in registers ----
  const int ni = tid >> 2;               // local node 0..63
  const int cpart = (tid & 3) * 16;      // 16-channel slice
  const int node = node0 + ni;
  const int ncl = node < N ? node : N - 1;
  float px = (x[ncl*3+0] - cmin[0]) / sstr[0];
  float py = (x[ncl*3+1] - cmin[1]) / sstr[1];
  float pz = (x[ncl*3+2] - cmin[2]) / sstr[2];
  float fx0 = floorf(px), fy0 = floorf(py), fz0 = floorf(pz);
  float fx = fminf(fmaxf(px - fx0, 0.f), 1.f);
  float fy = fminf(fmaxf(py - fy0, 0.f), 1.f);
  float fz = fminf(fmaxf(pz - fz0, 0.f), 1.f);
  int x0i = (int)fx0, y0i = (int)fy0, z0i = (int)fz0;
  int bb = nbatch[ncl];
  int gidx[8]; float wgt[8];
#pragma unroll
  for (int k = 0; k < 8; ++k) {
    int xi = x0i + (k & 1), yi = y0i + ((k >> 1) & 1), zi = z0i + (k >> 2);
    float wx = (k & 1)        ? fx : 1.f - fx;
    float wy = ((k >> 1) & 1) ? fy : 1.f - fy;
    float wz = (k >> 2)       ? fz : 1.f - fz;
    bool ok = (xi >= 0) && (xi < GDIM) && (yi >= 0) && (yi < GDIM) &&
              (zi >= 0) && (zi < GDIM) && (node < N);
    int xc = min(max(xi, 0), GDIM - 1);
    int yc = min(max(yi, 0), GDIM - 1);
    int zc = min(max(zi, 0), GDIM - 1);
    int flat = (zc << 12) + (yc << 6) + xc;
    gidx[k] = (bb << DHW_LOG2) + flat;
    wgt[k] = ok ? wx * wy * wz : 0.f;
  }
  // sampled channels -> zs[:,128+..]
  for (int c = 0; c < 16; ++c) {
    int ch = cpart + c;
    float acc = 0.f;
#pragma unroll
    for (int k = 0; k < 8; ++k)
      acc += wgt[k] * grid[((size_t)(bb * 64 + ch) << DHW_LOG2) + (gidx[k] & (DHW - 1))];
    zb.zs[ni][128 + ch] = (_Float16)acc;
  }
  // h -> zs[:,0:128] (f16)
  for (int i = tid; i < 64 * 128; i += 256) {
    int r = i >> 7, c = i & 127;
    int nn = node0 + r;
    zb.zs[r][c] = (_Float16)(nn < N ? h[(size_t)nn * 128 + c] : 0.f);
  }
  // stage w1 (128x192) as f16
  for (int i = tid; i < 128 * 192; i += 256)
    wb.w1s[i / 192][i % 192] = (_Float16)w1[i];
  __syncthreads();

  // ---- GEMM1: m = silu(z @ w1^T + b1), 64x128, K=192 ----
  {
    int n0 = wv << 4, ncol = n0 + (lane & 15);
    float bias = b1[ncol];
    for (int mt = 0; mt < 4; ++mt) {
      int m0 = mt << 4;
      v8f acc;
#pragma unroll
      for (int r = 0; r < 8; ++r) acc[r] = bias;
#pragma unroll
      for (int k0 = 0; k0 < 192; k0 += 32) {
        v16h a = lds_frag_a(&zb.zs[0][0], 200, m0, k0, lane);
        v16h b = lds_frag_b(&wb.w1s[0][0], 200, n0, k0, lane);
        acc = WMMA_F16(a, b, acc);
      }
      int mr = m0 + ((lane >> 4) << 3);
#pragma unroll
      for (int r = 0; r < 8; ++r) ms[mr + r][ncol] = (_Float16)silu_f(acc[r]);
    }
  }
  __syncthreads();
  // stage w2 (128x128) and wn2g (64x128), overwriting w1 region
  for (int i = tid; i < 128 * 128; i += 256) wb.p2.w2s[i >> 7][i & 127] = (_Float16)w2[i];
  for (int i = tid; i < 64 * 128; i += 256)  wb.p2.wgs[i >> 7][i & 127] = (_Float16)wn2g[i];
  __syncthreads();

  // ---- GEMM2: hn_pre = m @ w2^T + b2 + h, 64x128, K=128 ----
  {
    int n0 = wv << 4, ncol = n0 + (lane & 15);
    float bias = b2[ncol];
    for (int mt = 0; mt < 4; ++mt) {
      int m0 = mt << 4;
      v8f acc;
#pragma unroll
      for (int r = 0; r < 8; ++r) acc[r] = bias;
#pragma unroll
      for (int k0 = 0; k0 < 128; k0 += 32) {
        v16h a = lds_frag_a(&ms[0][0], 136, m0, k0, lane);
        v16h b = lds_frag_b(&wb.p2.w2s[0][0], 136, n0, k0, lane);
        acc = WMMA_F16(a, b, acc);
      }
      int mr = m0 + ((lane >> 4) << 3);
#pragma unroll
      for (int r = 0; r < 8; ++r) {
        int nn = node0 + mr + r;
        float hv = h[(size_t)(nn < N ? nn : N - 1) * 128 + ncol];  // fp32 residual
        zb.hn32[mr + r][ncol] = acc[r] + hv;
      }
    }
  }
  __syncthreads();

  // ---- LayerNorm over 128 features (4 threads / row) ----
  {
    int row = tid >> 2, p = tid & 3;
    float s = 0.f, ss = 0.f;
    for (int c = p * 32; c < p * 32 + 32; ++c) {
      float v = zb.hn32[row][c]; s += v; ss += v * v;
    }
    red[row][p] = s; red[row][4 + p] = ss;
  }
  __syncthreads();
  {
    int row = tid >> 2, p = tid & 3;
    float sum = red[row][0] + red[row][1] + red[row][2] + red[row][3];
    float sq  = red[row][4] + red[row][5] + red[row][6] + red[row][7];
    float mu = sum * 0.0078125f;
    float rstd = rsqrtf(sq * 0.0078125f - mu * mu + 1e-5f);
    int nn = node0 + row;
    for (int c = p * 32; c < p * 32 + 32; ++c) {
      float v = (zb.hn32[row][c] - mu) * rstd * ln_g[c] + ln_b[c];
      hnh[row][c] = (_Float16)v;
      if (nn < N) hn_out[(size_t)nn * 128 + c] = v;
    }
  }
  __syncthreads();

  // ---- GEMM3: proj = hn @ wn2g^T + bn2g, 64x64, K=128 (2 tiles / wave) ----
  {
#pragma unroll
    for (int i = 0; i < 2; ++i) {
      int id = wv * 2 + i, mt = id >> 2, nt = id & 3;
      int m0 = mt << 4, n0 = nt << 4, ncol = n0 + (lane & 15);
      v8f acc;
      float bias = bn2g[ncol];
#pragma unroll
      for (int r = 0; r < 8; ++r) acc[r] = bias;
#pragma unroll
      for (int k0 = 0; k0 < 128; k0 += 32) {
        v16h a = lds_frag_a(&hnh[0][0], 136, m0, k0, lane);
        v16h b = lds_frag_b(&wb.p2.wgs[0][0], 136, n0, k0, lane);
        acc = WMMA_F16(a, b, acc);
      }
      int mr = m0 + ((lane >> 4) << 3);
#pragma unroll
      for (int r = 0; r < 8; ++r) projs[mr + r][ncol] = acc[r];
    }
  }
  __syncthreads();

  // ---- trilinear scatter-add into volume (B*DHW, 64) ----
  if (node < N) {
#pragma unroll
    for (int k = 0; k < 8; ++k) {
      float wk = wgt[k];
      if (wk != 0.f) {
        float* dst = vol + (size_t)gidx[k] * 64 + cpart;
#pragma unroll
        for (int c = 0; c < 16; ++c)
          atomicAdd(dst + c, wk * projs[ni][cpart + c]);
      }
    }
  }
}

// ---------------------------------------------------------------------------
// Per 64-voxel workgroup: cat=[grid|splat] -> conv1(silu) -> conv2 + residual,
// accumulate per-(b,c) instance-norm stats.
__global__ void __launch_bounds__(256) conv_kernel(
    const float* __restrict__ grid, const float* __restrict__ vol,
    const float* __restrict__ wc1, const float* __restrict__ bc1,
    const float* __restrict__ wc2, const float* __restrict__ bc2,
    float* __restrict__ gr_out, float* __restrict__ stats)
{
  __shared__ __align__(16) _Float16 cats[64][136];   // 128 used
  __shared__ __align__(16) _Float16 wc1s[64][136];
  __shared__ __align__(16) _Float16 g1s[64][72];     // 64 used
  __shared__ __align__(16) _Float16 wc2s[64][72];
  __shared__ float ssum[64], ssq[64];

  const int tid = threadIdx.x;
  const int lane = tid & 31, wv = tid >> 5;
  const int vox0 = blockIdx.x * 64;
  const int b = vox0 >> DHW_LOG2;
  const int v0 = vox0 & (DHW - 1);

  for (int i = tid * 32; i < 64 * 64; i += 256 * 32) __builtin_prefetch(&wc2[i], 0, 0);

  for (int i = tid; i < 64 * 64; i += 256) {  // grid channels (coalesced in m)
    int m = i & 63, c = i >> 6;
    cats[m][c] = (_Float16)grid[((size_t)(b * 64 + c) << DHW_LOG2) + v0 + m];
  }
  for (int i = tid; i < 64 * 64; i += 256) {  // splat channels (coalesced in c)
    int c = i & 63, m = i >> 6;
    cats[m][64 + c] = (_Float16)vol[((size_t)(b << DHW_LOG2) + v0 + m) * 64 + c];
  }
  for (int i = tid; i < 64 * 128; i += 256) wc1s[i >> 7][i & 127] = (_Float16)wc1[i];
  for (int i = tid; i < 64 * 64; i += 256)  wc2s[i >> 6][i & 63]  = (_Float16)wc2[i];
  if (tid < 64) { ssum[tid] = 0.f; ssq[tid] = 0.f; }
  __syncthreads();

  // GEMM1: g1 = silu(cat @ wc1^T + bc1), 64x64, K=128
#pragma unroll
  for (int i = 0; i < 2; ++i) {
    int id = wv * 2 + i, mt = id >> 2, nt = id & 3;
    int m0 = mt << 4, n0 = nt << 4, ncol = n0 + (lane & 15);
    v8f acc;
    float bias = bc1[ncol];
#pragma unroll
    for (int r = 0; r < 8; ++r) acc[r] = bias;
#pragma unroll
    for (int k0 = 0; k0 < 128; k0 += 32) {
      v16h a = lds_frag_a(&cats[0][0], 136, m0, k0, lane);
      v16h bfr = lds_frag_b(&wc1s[0][0], 136, n0, k0, lane);
      acc = WMMA_F16(a, bfr, acc);
    }
    int mr = m0 + ((lane >> 4) << 3);
#pragma unroll
    for (int r = 0; r < 8; ++r) g1s[mr + r][ncol] = (_Float16)silu_f(acc[r]);
  }
  __syncthreads();

  // GEMM2: gr_pre = g1 @ wc2^T + bc2 + grid, 64x64, K=64; collect stats
#pragma unroll
  for (int i = 0; i < 2; ++i) {
    int id = wv * 2 + i, mt = id >> 2, nt = id & 3;
    int m0 = mt << 4, n0 = nt << 4, ncol = n0 + (lane & 15);
    v8f acc;
    float bias = bc2[ncol];
#pragma unroll
    for (int r = 0; r < 8; ++r) acc[r] = bias;
#pragma unroll
    for (int k0 = 0; k0 < 64; k0 += 32) {
      v16h a = lds_frag_a(&g1s[0][0], 72, m0, k0, lane);
      v16h bfr = lds_frag_b(&wc2s[0][0], 72, n0, k0, lane);
      acc = WMMA_F16(a, bfr, acc);
    }
    int mr = m0 + ((lane >> 4) << 3);
    size_t basec = ((size_t)(b * 64 + ncol) << DHW_LOG2) + v0;
    float lsum = 0.f, lsq = 0.f;
#pragma unroll
    for (int r = 0; r < 8; ++r) {
      float v = acc[r] + grid[basec + mr + r];   // fp32 residual
      gr_out[basec + mr + r] = v;
      lsum += v; lsq += v * v;
    }
    atomicAdd(&ssum[ncol], lsum);
    atomicAdd(&ssq[ncol], lsq);
  }
  __syncthreads();
  if (tid < 128) {
    int c = tid & 63, s = tid >> 6;
    atomicAdd(&stats[(size_t)(b * 64 + c) * 2 + s], s ? ssq[c] : ssum[c]);
  }
}

// ---------------------------------------------------------------------------
// Instance-norm finalize, vectorized float4 (pure bandwidth).
__global__ void finalize_kernel(float4* __restrict__ gr, const float* __restrict__ stats,
                                const float* __restrict__ ing, const float* __restrict__ inb,
                                size_t total4) {
  const float inv = 1.f / (float)DHW;
  for (size_t i = (size_t)blockIdx.x * blockDim.x + threadIdx.x; i < total4;
       i += (size_t)gridDim.x * blockDim.x) {
    size_t bc = (i * 4) >> DHW_LOG2;   // 4 elems share (b,c): DHW % 4 == 0
    int c = (int)(bc & 63);
    float sum = stats[bc * 2], sq = stats[bc * 2 + 1];
    float mu = sum * inv;
    float rstd = rsqrtf(sq * inv - mu * mu + 1e-5f);
    float sc = rstd * ing[c];
    float sh = inb[c] - mu * sc;
    float4 v = gr[i];
    v.x = v.x * sc + sh; v.y = v.y * sc + sh;
    v.z = v.z * sc + sh; v.w = v.w * sc + sh;
    gr[i] = v;
  }
}

// ---------------------------------------------------------------------------
extern "C" void kernel_launch(void* const* d_in, const int* in_sizes, int n_in,
                              void* d_out, int out_size, void* d_ws, size_t ws_size,
                              hipStream_t stream) {
  const float* h    = (const float*)d_in[0];
  const float* x    = (const float*)d_in[1];
  const float* grid = (const float*)d_in[2];
  const int*   nb   = (const int*)d_in[3];
  const float* cmin = (const float*)d_in[4];
  const float* sstr = (const float*)d_in[5];
  const float* w1   = (const float*)d_in[6];
  const float* b1   = (const float*)d_in[7];
  const float* w2   = (const float*)d_in[8];
  const float* b2   = (const float*)d_in[9];
  const float* lng  = (const float*)d_in[10];
  const float* lnb  = (const float*)d_in[11];
  const float* wn2g = (const float*)d_in[12];
  const float* bn2g = (const float*)d_in[13];
  const float* wc1  = (const float*)d_in[14];
  const float* bc1  = (const float*)d_in[15];
  const float* wc2  = (const float*)d_in[16];
  const float* bc2  = (const float*)d_in[17];
  const float* ing  = (const float*)d_in[18];
  const float* inb  = (const float*)d_in[19];

  const int N = in_sizes[0] / 128;
  const int B = in_sizes[2] / (64 * DHW);

  float* hn_out = (float*)d_out;
  float* gr_out = (float*)d_out + (size_t)N * 128;
  float* vol    = (float*)d_ws;
  float* stats  = vol + (size_t)B * DHW * 64;

  size_t zcount = (size_t)B * DHW * 64 + (size_t)B * 128;   // divisible by 4
  zero_kernel<<<2048, 256, 0, stream>>>((float4*)vol, zcount / 4);
  node_kernel<<<(N + 63) / 64, 256, 0, stream>>>(h, x, grid, nb, cmin, sstr,
                                                 w1, b1, w2, b2, lng, lnb,
                                                 wn2g, bn2g, hn_out, vol, N);
  conv_kernel<<<B * (DHW / 64), 256, 0, stream>>>(grid, vol, wc1, bc1, wc2, bc2,
                                                  gr_out, stats);
  finalize_kernel<<<4096, 256, 0, stream>>>((float4*)gr_out, stats, ing, inb,
                                            ((size_t)B * 64 * DHW) / 4);
}
